// Conv4d_17411797418345
// MI455X (gfx1250) — matrix operand order repaired and checked by reading, not compiled, not run on hardware
//
#include <hip/hip_runtime.h>
#include <hip/hip_bf16.h>

typedef __attribute__((ext_vector_type(16))) _Float16 v16h;
typedef __attribute__((ext_vector_type(8)))  float    v8f;

#define DD    32
#define IC    16
#define OC    32
#define TAPS  81          // 3^4 kernel taps
#define CHUNKS 42         // 41 real K32-chunks + 1 zero chunk -> even for unroll-2

static constexpr int D2 = DD * DD;           // 1024
static constexpr int D3 = DD * DD * DD;      // 32768
static constexpr int D4 = DD * DD * DD * DD; // 1048576

// ---------------------------------------------------------------------
// Compile-time tap table: packed (k1,k2,k3,k4) and linearized spatial
// delta (k1-1)*D3 + (k2-1)*D2 + (k3-1)*DD + (k4-1).  Entries 81..83 are
// sentinels (k=100) that fail every bounds check -> zero padding for the
// odd tail tap and for the extra even-ing chunk.
// ---------------------------------------------------------------------
struct TapTab { int kpack[84]; int delta[84]; };
static constexpr TapTab makeTab() {
    TapTab t{};
    for (int tap = 0; tap < 84; ++tap) {
        if (tap < 81) {
            int k1 = tap / 27, r = tap % 27;
            int k2 = r / 9;
            int k3 = (r % 9) / 3;
            int k4 = r % 3;
            t.kpack[tap] = k1 | (k2 << 8) | (k3 << 16) | (k4 << 24);
            t.delta[tap] = (k1 - 1) * D3 + (k2 - 1) * D2 + (k3 - 1) * DD + (k4 - 1);
        } else {
            t.kpack[tap] = 100 | (100 << 8) | (100 << 16) | (100 << 24);
            t.delta[tap] = 0;
        }
    }
    return t;
}
constexpr TapTab g_tab = makeTab();

// LDS layout:
//   sB: 84 tiles (42 chunks x 2 OC-halves) x 32 lanes x 16 halves  = 86,016 B
//   sC: 8 waves x 16 rows x 33 floats (stride-33 = conflict-free)  = 16,896 B
static constexpr int SB_HALVES = 84 * 32 * 16;
static constexpr int SC_FLOATS = 8 * 16 * 33;
static constexpr size_t SMEM_BYTES = (size_t)SB_HALVES * 2 + (size_t)SC_FLOATS * 4;

// Build one A fragment (16 positions x 32 K, f16) for K-chunk c.
// A layout (16-bit A 16x32): lane half h, element j -> K = (j<8?0:16)+h*8+(j&7)
//   => tap select ts = j>>3, ic = h*8 + (j&7).
//
// BRANCH-FREE: out-of-bounds taps load from a safe clamped index (x[jj*D4],
// always in-bounds, finite data) and the value is zeroed with v_cndmask.
// Channel loads are paired around 32-bit pair-bases laundered through an
// empty asm so every load keeps the cheap saddr+voffset+imm(0/4MiB) form
// instead of 64-bit carry-chain addressing for jj>=2.
__device__ __forceinline__ v16h load_a(const float* __restrict__ x, int c,
                                       int o1, int o2, int o3, int o4, int bb)
{
    v16h a;
#pragma unroll
    for (int ts = 0; ts < 2; ++ts) {
        int tap = 2 * c + ts;
        int kp  = g_tab.kpack[tap];     // uniform -> s_load
        int dl  = g_tab.delta[tap];     // uniform -> s_load
        int k1 = kp & 255, k2 = (kp >> 8) & 255;
        int k3 = (kp >> 16) & 255, k4 = (kp >> 24) & 255;
        // Bitwise & (no short-circuit): uniform part folds to SALU + one
        // per-lane compare (clang OR-combines the range checks).
        int vu = (int)((unsigned)(o1 + k1 - 1) < DD) &
                 (int)((unsigned)(o2 + k2 - 1) < DD) &
                 (int)((unsigned)(o3 + k3 - 1) < DD);
        bool valid = (vu & (int)((unsigned)(o4 + k4 - 1) < DD)) != 0;
        int base = valid ? (bb + dl) : 0;   // safe index select (one cndmask)
        float raw[8];
#pragma unroll
        for (int q = 0; q < 4; ++q) {
            int bq = base + 2 * q * D4;     // 32-bit pair base
            asm("" : "+v"(bq));             // opaque: block 64-bit re-association
            raw[2 * q]     = x[bq];         // imm offset 0
            raw[2 * q + 1] = x[bq + D4];    // imm offset 4 MiB (in range)
        }
#pragma unroll
        for (int jj = 0; jj < 8; ++jj) {    // value select + convert (no branch)
            float v = valid ? raw[jj] : 0.0f;
            a[ts * 8 + jj] = (_Float16)v;
        }
    }
    return a;
}

__global__ __launch_bounds__(256)
void conv4d_wmma_f16(const float* __restrict__ x,
                     const float* __restrict__ w,
                     const float* __restrict__ bias,
                     float* __restrict__ out)
{
    extern __shared__ _Float16 smem[];
    _Float16* sB = smem;                                  // [84][32][16] halves
    float*    sC = (float*)(smem + SB_HALVES);            // [8][16][33] floats

    const int tid  = threadIdx.x;
    const int lane = tid & 31;
    const int wave = tid >> 5;
    const int h    = lane >> 4;     // half-wave id (selects K/ic sub-range)
    const int mlan = lane & 15;     // M coordinate inside tile / N coord for B,C

    // ---------------------------------------------------------------
    // Stage all weights into LDS in the WMMA B-fragment layout.
    // B tile (K=32 x N=16, f16): lane half h, element j ->
    // K = (j<8?0:16) + h*8 + (j&7) (matches A layout).
    // K within a chunk = ts*16 + ic  (ts = tap select, ic = input chan).
    // ---------------------------------------------------------------
    for (int idx = tid; idx < SB_HALVES; idx += 256) {
        int j       = idx & 15;          // half index within lane fragment
        int l       = (idx >> 4) & 31;   // lane
        int tileIdx = idx >> 9;          // 0..83 = chunk*2 + ocHalf
        int c       = tileIdx >> 1;
        int t       = tileIdx & 1;
        int K       = (j < 8 ? 0 : 16) + ((l >> 4) << 3) + (j & 7); // 0..31
        int tap     = 2 * c + (K >> 4);
        int ic      = K & 15;
        int n       = (l & 15) + 16 * t;         // output channel
        float v = (tap < TAPS) ? w[n * (IC * TAPS) + ic * TAPS + tap] : 0.0f;
        sB[idx] = (_Float16)v;
    }
    __syncthreads();

    // ---------------------------------------------------------------
    // Each wave owns one M-tile: 16 consecutive o4 positions at fixed
    // (o1,o2,o3), and all 32 output channels (two accumulators).
    // Wave-uniform scalars are forced into SGPRs via readfirstlane.
    // ---------------------------------------------------------------
    int mtile  = blockIdx.x * 8 + wave;          // 0..65535 (uniform per wave)
    int o4base = __builtin_amdgcn_readfirstlane((mtile & 1) * 16);
    int rest   = __builtin_amdgcn_readfirstlane(mtile >> 1);
    int o3 = rest & 31;
    int o2 = (rest >> 5) & 31;
    int o1 = rest >> 10;
    const int o4 = o4base + mlan;                // per-lane

    // Per-lane 32-bit base element index for A loads (ic = h*8 chan base).
    const int su = o1 * D3 + o2 * D2 + o3 * DD;  // uniform spatial base
    const int bb = h * 8 * D4 + su + o4;

    v8f acc0 = {};   // OC 0..15
    v8f acc1 = {};   // OC 16..31

    // Software pipeline, ping-pong A fragments (unroll-2):
    // chunk c's WMMAs only wait on loads issued one iteration earlier.
    v16h aP = load_a(x, 0, o1, o2, o3, o4, bb);

    for (int c = 0; c < CHUNKS; c += 2) {
        const v16h* bp = (const v16h*)(sB + (size_t)c * 1024);
        v16h b0 = bp[lane];
        v16h b1 = bp[32 + lane];
        v16h aQ = load_a(x, c + 1, o1, o2, o3, o4, bb);

        acc0 = __builtin_amdgcn_wmma_f32_16x16x32_f16(
            false, aP, false, b0, (short)0, acc0, false, false);
        acc1 = __builtin_amdgcn_wmma_f32_16x16x32_f16(
            false, aP, false, b1, (short)0, acc1, false, false);

        v16h b2 = bp[64 + lane];
        v16h b3 = bp[96 + lane];
        int cn = (c + 2 < CHUNKS) ? c + 2 : 0;   // last prefetch is a dummy
        aP = load_a(x, cn, o1, o2, o3, o4, bb);

        acc0 = __builtin_amdgcn_wmma_f32_16x16x32_f16(
            false, aQ, false, b2, (short)0, acc0, false, false);
        acc1 = __builtin_amdgcn_wmma_f32_16x16x32_f16(
            false, aQ, false, b3, (short)0, acc1, false, false);
    }

    // ---------------------------------------------------------------
    // C layout: lane -> N = lane&15 (+16 for acc1), VGPR r -> M = r + 8*h.
    // Transpose through LDS (stride 33 -> conflict-free) so the 128 MiB
    // of output stores are coalesced 64B runs along o4.
    // ---------------------------------------------------------------
    float* sCw = sC + wave * (16 * 33);
#pragma unroll
    for (int r = 0; r < 8; ++r) {
        int m = r + 8 * h;
        sCw[m * 33 + mlan]      = acc0[r];
        sCw[m * 33 + mlan + 16] = acc1[r];
    }
    // Intra-wave LDS RAW: DS ops are in-order per wave; make the
    // dependency explicit with the CDNA5 split DS counter.
    asm volatile("s_wait_dscnt 0" ::: "memory");

    int outSpatial = su + o4base;
#pragma unroll
    for (int i = 0; i < 16; ++i) {
        int oc = i * 2 + h;            // each half-wave writes one channel row
        float v = sCw[mlan * 33 + oc] + bias[oc];
        out[oc * D4 + outSpatial + mlan] = v;
    }
}

extern "C" void kernel_launch(void* const* d_in, const int* in_sizes, int n_in,
                              void* d_out, int out_size, void* d_ws, size_t ws_size,
                              hipStream_t stream) {
    (void)in_sizes; (void)n_in; (void)out_size; (void)d_ws; (void)ws_size;
    const float* x    = (const float*)d_in[0];
    const float* w    = (const float*)d_in[1];
    const float* bias = (const float*)d_in[2];
    float* out        = (float*)d_out;

    // >64KB dynamic LDS opt-in (idempotent; host-side, graph-capture safe)
    (void)hipFuncSetAttribute((const void*)conv4d_wmma_f16,
                              hipFuncAttributeMaxDynamicSharedMemorySize,
                              (int)SMEM_BYTES);

    // M-tiles: 32^4 / 16 = 65536; 8 waves (one tile each) per 256-thread block
    dim3 grid(65536 / 8), block(256);
    conv4d_wmma_f16<<<grid, block, SMEM_BYTES, stream>>>(x, w, bias, out);
}
